// MCLSTM_59768764891409
// MI455X (gfx1250) — matrix operand order repaired and checked by reading, not compiled, not run on hardware
//
#include <hip/hip_runtime.h>
#include <hip/hip_bf16.h>

#define HS   128
#define BSZ  256
#define SEQL 512
#define INSZ 64
#define NG   517        // HS*4 + 5
#define GP   528        // padded to 33*16
#define USTR 136        // padded K-stride (f16) for transposed U in LDS (16B aligned)
#define WSTR 72         // padded K-stride (f16) for transposed W in LDS (16B aligned)
#define HSTR 136        // h_lds row stride (f16, 16B aligned)

typedef _Float16 v16h __attribute__((ext_vector_type(16)));
typedef float    v8f  __attribute__((ext_vector_type(8)));

union Frag16 {
  v16h v;
  unsigned int u[8];
  uint4 q[2];
  _Float16 e[16];
};

__device__ __forceinline__ float fast_sigmoid(float x) {
  return 1.0f / (1.0f + __expf(-x));
}
__device__ __forceinline__ float fast_tanh(float x) {
  x = fminf(fmaxf(x, -30.0f), 30.0f);
  float e = __expf(2.0f * x);
  return (e - 1.0f) / (e + 1.0f);
}

__device__ __forceinline__ v8f wmma_f16(v16h a, v16h b, v8f c) {
  // D = A(16x32 f16) * B(32x16 f16) + C(16x16 f32)
  return __builtin_amdgcn_wmma_f32_16x16x32_f16(false, a, false, b, (short)0, c,
                                                false, false);
}

// A fragment (16x32 f16): row M = lane&15 (both lane halves), hi = lane>=16.
// VGPR v holds K = kbase + 2*(v&3) + ((v>>2)<<4) + hi*8 (packed f16 pair).
__device__ __forceinline__ v16h load_a_global_f32(const float* row, int kbase, int hi) {
  Frag16 f;
#pragma unroll
  for (int v = 0; v < 8; ++v) {
    int k = kbase + 2 * (v & 3) + ((v >> 2) << 4) + (hi << 3);
    float2 p = *(const float2*)(row + k);
    f.e[2 * v]     = (_Float16)p.x;
    f.e[2 * v + 1] = (_Float16)p.y;
  }
  return f.v;
}

// Same A layout from LDS f16: dwords v=0..3 and v=4..7 are two contiguous 16B runs.
__device__ __forceinline__ v16h load_a_lds_f16(const _Float16* hl, int m, int kbase, int hi) {
  Frag16 f;
  const _Float16* p = hl + m * HSTR + kbase + (hi << 3);
  f.q[0] = *(const uint4*)(p);        // K = kbase+0..7   (this lane half)
  f.q[1] = *(const uint4*)(p + 16);   // K = kbase+16..23 (this lane half)
  return f.v;
}

// B fragment (32x16 f16): col N = lane&15; same K layout as A:
// VGPR v holds K pair (kbase + 2*(v&3) + ((v>>2)<<4) + hi*8).
// mat is column-major per g (element (k,g) at mat[g*stride + k]) so the whole
// per-lane fragment is two 16B runs (K +0..7 and K +16..23) -> two ds_load_b128.
__device__ __forceinline__ v16h load_b_lds_f16(const _Float16* mat, int stride,
                                               int gcol, int kbase, int hi) {
  Frag16 f;
  const _Float16* p = mat + gcol * stride + kbase + (hi << 3);
  f.q[0] = *(const uint4*)(p);
  f.q[1] = *(const uint4*)(p + 16);
  return f.v;
}

__global__ void __launch_bounds__(288, 1)
mclstm_persistent(const float* __restrict__ x, const float* __restrict__ W,
                  const float* __restrict__ U, const float* __restrict__ k1,
                  const float* __restrict__ bias, const float* __restrict__ bias1,
                  float* __restrict__ out)
{
  extern __shared__ char smem_raw[];
  _Float16* U_t = (_Float16*)smem_raw;                 // [GP][USTR] f16, transposed U
  _Float16* W_t = U_t + (size_t)GP * USTR;             // [GP][WSTR] f16, transposed W
  _Float16* h_l = W_t + (size_t)GP * WSTR;             // [16][HSTR] f16 hidden state
  float* d_raw  = (float*)(h_l + 16 * HSTR);           // [16][8]
  float* d_sm   = d_raw + 16 * 8;                      // [16][8]

  const int tid  = threadIdx.x;
  const int wave = tid >> 5;      // 0..8
  const int lane = tid & 31;
  const int hi   = lane >> 4;
  const int lc   = lane & 15;
  const int b0   = blockIdx.x * 16;

  // ---- one-time cooperative transpose of U (128xNG) / W (64xNG) into LDS (f16) ----
  for (int idx = tid; idx < GP * HS; idx += blockDim.x) {
    int g = idx >> 7, i = idx & 127;
    float v = (g < NG) ? U[(size_t)i * NG + g] : 0.0f;
    U_t[g * USTR + i] = (_Float16)v;
  }
  for (int idx = tid; idx < GP * INSZ; idx += blockDim.x) {
    int g = idx >> 6, i = idx & 63;
    float v = (g < NG) ? W[(size_t)i * NG + g] : 0.0f;
    W_t[g * WSTR + i] = (_Float16)v;
  }
  for (int idx = tid; idx < 16 * HSTR; idx += blockDim.x) h_l[idx] = (_Float16)0.0f;
  __syncthreads();

  const size_t TS = (size_t)BSZ * SEQL * HS;
  float* outH  = out;
  float* outC1 = out + TS;
  float* outC2 = out + 2 * TS;
  float* outC3 = out + 3 * TS;
  float* outC4 = out + 4 * TS;
  float* outC5 = out + 5 * TS;
  float* outHF = out + 6 * TS;
  float* outCF = outHF + (size_t)BSZ * HS;
  float* outDS = outCF + (size_t)BSZ * HS;

  const float* xrow = x + (size_t)(b0 + lc) * SEQL * INSZ;

  // per-lane constants
  const int j0 = (wave < 8) ? wave * 16 : 0;
  const int j  = j0 + lc;
  float bI = 0, bF = 0, bG = 0, bO = 0, s1 = 0, s2 = 0;
  float kw0 = 0, kw1 = 0, kw2 = 0, kw3 = 0, kw4 = 0, bD = 0;
  if (wave < 8) {
    bI = bias[j]; bF = bias[HS + j]; bG = bias[2 * HS + j]; bO = bias[3 * HS + j];
    s1 = bias1[j]; s2 = bias1[HS + j];
    kw0 = k1[j]; kw1 = k1[HS + j]; kw2 = k1[2 * HS + j];
    kw3 = k1[3 * HS + j]; kw4 = k1[4 * HS + j];
  } else {
    bD = (lc < 5) ? bias[4 * HS + lc] : 0.0f;
  }

  v8f c_state;
#pragma unroll
  for (int r = 0; r < 8; ++r) c_state[r] = 0.0f;

  for (int t = 0; t < SEQL; ++t) {
    __syncthreads();  // barrier A: previous step's h_l writes visible

    const float* xr = xrow + (size_t)t * INSZ;
    v8f ai, af, ag, ao;
    v8f it, ft, gt, ot;

    if (wave < 8) {
#pragma unroll
      for (int r = 0; r < 8; ++r) { ai[r] = bI; af[r] = bF; ag[r] = bG; ao[r] = bO; }
      if (t + 1 < SEQL) __builtin_prefetch(xr + INSZ, 0, 0);
      // gates += x_t @ W   (K = 64)
#pragma unroll
      for (int kb = 0; kb < 2; ++kb) {
        v16h A = load_a_global_f32(xr, kb * 32, hi);
        ai = wmma_f16(A, load_b_lds_f16(W_t, WSTR, 0 * HS + j, kb * 32, hi), ai);
        af = wmma_f16(A, load_b_lds_f16(W_t, WSTR, 1 * HS + j, kb * 32, hi), af);
        ag = wmma_f16(A, load_b_lds_f16(W_t, WSTR, 2 * HS + j, kb * 32, hi), ag);
        ao = wmma_f16(A, load_b_lds_f16(W_t, WSTR, 3 * HS + j, kb * 32, hi), ao);
      }
      // gates += h @ U   (K = 128)
#pragma unroll
      for (int kb = 0; kb < 4; ++kb) {
        v16h A = load_a_lds_f16(h_l, lc, kb * 32, hi);
        ai = wmma_f16(A, load_b_lds_f16(U_t, USTR, 0 * HS + j, kb * 32, hi), ai);
        af = wmma_f16(A, load_b_lds_f16(U_t, USTR, 1 * HS + j, kb * 32, hi), af);
        ag = wmma_f16(A, load_b_lds_f16(U_t, USTR, 2 * HS + j, kb * 32, hi), ag);
        ao = wmma_f16(A, load_b_lds_f16(U_t, USTR, 3 * HS + j, kb * 32, hi), ao);
      }
    } else {
      // wave 8: d-gate tile (cols 512..516, padded to 528 with zeroed weights)
      v8f ad;
#pragma unroll
      for (int r = 0; r < 8; ++r) ad[r] = bD;
      const int gd = 4 * HS + lc;
#pragma unroll
      for (int kb = 0; kb < 2; ++kb) {
        v16h A = load_a_global_f32(xr, kb * 32, hi);
        ad = wmma_f16(A, load_b_lds_f16(W_t, WSTR, gd, kb * 32, hi), ad);
      }
#pragma unroll
      for (int kb = 0; kb < 4; ++kb) {
        v16h A = load_a_lds_f16(h_l, lc, kb * 32, hi);
        ad = wmma_f16(A, load_b_lds_f16(U_t, USTR, gd, kb * 32, hi), ad);
      }
      if (lc < 5) {
#pragma unroll
        for (int r = 0; r < 8; ++r) {
          int row = r + 8 * hi;
          d_raw[row * 8 + lc] = fast_tanh(ad[r]);
        }
      }
    }

    __syncthreads();  // barrier B: d_raw ready; all h_l reads done

    if (wave == 8 && lane < 16) {
      const int row = lane;
      float4 vlo = *(const float4*)(d_raw + row * 8);
      float v4 = d_raw[row * 8 + 4];
      float m = fmaxf(fmaxf(fmaxf(vlo.x, vlo.y), fmaxf(vlo.z, vlo.w)), v4);
      float e0 = __expf(vlo.x - m), e1 = __expf(vlo.y - m), e2 = __expf(vlo.z - m),
            e3 = __expf(vlo.w - m), e4 = __expf(v4 - m);
      float inv = 1.0f / (e0 + e1 + e2 + e3 + e4);
      float p0 = e0 * inv, p1 = e1 * inv, p2 = e2 * inv, p3 = e3 * inv, p4 = e4 * inv;
      *(float4*)(d_sm + row * 8) = make_float4(p0, p1, p2, p3);
      d_sm[row * 8 + 4] = p4;
      if (t == SEQL - 1) {
        size_t o = (size_t)(b0 + row) * 5;
        outDS[o + 0] = p0; outDS[o + 1] = p1; outDS[o + 2] = p2;
        outDS[o + 3] = p3; outDS[o + 4] = p4;
      }
    }

    if (wave < 8) {
#pragma unroll
      for (int r = 0; r < 8; ++r) {
        it[r] = fast_sigmoid(ai[r]);
        ft[r] = fast_sigmoid(af[r]);
        gt[r] = fast_tanh(ag[r]);
        ot[r] = fast_sigmoid(ao[r]);
      }
    }

    __syncthreads();  // barrier C: d_sm ready

    if (wave < 8) {
#pragma unroll
      for (int r = 0; r < 8; ++r) {
        const int row = r + 8 * hi;
        const int b   = b0 + row;
        float c5 = c_state[r];
        float c1 = fast_tanh(gt[r]);
        float c3 = ft[r] * c5 + it[r] * gt[r];
        float c2 = s2 * c3 + (1.0f - s2) * c1;
        float c4 = s1 * c3 + (1.0f - s1) * c5;
        float4 dlo = *(const float4*)(d_sm + row * 8);
        float d4 = d_sm[row * 8 + 4];
        float cn = dlo.x * kw0 * c1 + dlo.y * kw1 * c2 + dlo.z * kw2 * c3 +
                   dlo.w * kw3 * c4 + d4 * kw4 * c5;
        float hn = ot[r] * fast_tanh(cn);
        c_state[r] = cn;
        size_t base = ((size_t)b * SEQL + t) * HS + j;
        outH [base] = hn;
        outC1[base] = c1;
        outC2[base] = c2;
        outC3[base] = c3;
        outC4[base] = c4;
        outC5[base] = c5;
        h_l[row * HSTR + j] = (_Float16)hn;
        if (t == SEQL - 1) {
          outHF[(size_t)b * HS + j] = hn;
          outCF[(size_t)b * HS + j] = cn;
        }
      }
    }
  }
}

extern "C" void kernel_launch(void* const* d_in, const int* in_sizes, int n_in,
                              void* d_out, int out_size, void* d_ws, size_t ws_size,
                              hipStream_t stream) {
  (void)in_sizes; (void)n_in; (void)d_ws; (void)ws_size; (void)out_size;
  const float* x     = (const float*)d_in[0];
  const float* W     = (const float*)d_in[1];
  const float* U     = (const float*)d_in[2];
  const float* k1    = (const float*)d_in[3];
  const float* bias  = (const float*)d_in[4];
  const float* bias1 = (const float*)d_in[5];
  float* out = (float*)d_out;

  const size_t shmem = (size_t)GP * USTR * 2 + (size_t)GP * WSTR * 2 +
                       (size_t)16 * HSTR * 2 + 2 * 16 * 8 * sizeof(float);
  hipFuncSetAttribute((const void*)mclstm_persistent,
                      hipFuncAttributeMaxDynamicSharedMemorySize, (int)shmem);
  mclstm_persistent<<<dim3(BSZ / 16), dim3(288), shmem, stream>>>(
      x, W, U, k1, bias, bias1, out);
}
